// TSMixerH_14027363189073
// MI455X (gfx1250) — matrix operand order repaired and checked by reading, not compiled, not run on hardware
//
#include <hip/hip_runtime.h>
#include <hip/hip_bf16.h>

// Problem sizes (match reference)
#define B_N   32
#define C_N   256
#define L_N   336
#define O_N   96
#define K_N   4
#define NL_N  2
#define DFF_N 1024

// LDS row strides (padded to break 64-bank conflicts)
#define ZS 356    // floats per z row  (336 data + 16 zero-pad + 4 spare)
#define HS 1032   // halves per h row  (1024 data + 8 spare)

// WMMA tiling
#define NT1 64    // GEMM1 N-tiles (DFF/16)
#define KS1 11    // K-steps over L (336 -> 352 = 11*32, zero-padded)
#define NT2 21    // GEMM2 N-tiles (L/16)
#define KS2 32    // K-steps over DFF (1024 = 32*32)
#define NTH 6     // head N-tiles (O/16)

typedef __attribute__((ext_vector_type(16))) __bf16 v16bf;
typedef __attribute__((ext_vector_type(8)))  __bf16 v8bf;
typedef __attribute__((ext_vector_type(8)))  float  v8f;
typedef __attribute__((ext_vector_type(4)))  float  v4f;

// Swizzled-weight sizes in bf16 elements
#define W1_HALVES ((size_t)(K_N*NL_N) * NT1 * KS1 * 512)  // 2,883,584
#define W2_HALVES ((size_t)(K_N*NL_N) * NT2 * KS2 * 512)  // 2,752,512
#define WH_HALVES ((size_t)K_N * NTH * KS1 * 512)         //   135,168

// ---------------------------------------------------------------------------
// A-fragment loaders: ISA 16-bit A 16x32 layout.
//   lane <16 : M=lane,    halves 0-7 = K kb+0..7,  halves 8-15 = K kb+16..23
//   lane>=16 : M=lane-16, halves 0-7 = K kb+8..15, halves 8-15 = K kb+24..31
// Caller passes p = &row[kb + o1] with o1 = (lane>=16 ? 8 : 0).
// ---------------------------------------------------------------------------
__device__ inline v16bf loadA_f32(const float* p) {
  v4f x0 = *(const v4f*)(p);
  v4f x1 = *(const v4f*)(p + 4);
  v4f y0 = *(const v4f*)(p + 16);
  v4f y1 = *(const v4f*)(p + 20);
  v16bf a;
#pragma unroll
  for (int j = 0; j < 4; ++j) {
    a[j]      = (__bf16)x0[j];
    a[j + 4]  = (__bf16)x1[j];
    a[j + 8]  = (__bf16)y0[j];
    a[j + 12] = (__bf16)y1[j];
  }
  return a;
}

__device__ inline v16bf loadA_bf16(const __bf16* p) {
  v8bf lo = *(const v8bf*)(p);
  v8bf hi = *(const v8bf*)(p + 16);
  return __builtin_shufflevector(lo, hi, 0,1,2,3,4,5,6,7,8,9,10,11,12,13,14,15);
}

__device__ inline v8f wmma_bf16(v16bf a, v16bf b, v8f c) {
  return __builtin_amdgcn_wmma_f32_16x16x32_bf16(
      /*neg_a=*/false, a, /*neg_b=*/false, b,
      /*c_mod=*/(short)0, c, /*reuse_a=*/false, /*reuse_b=*/false);
}

// ---------------------------------------------------------------------------
// Weight pre-swizzle kernels: f32 weights -> bf16 in B-fragment order.
// Fragment element (frag, lane, j) holds B[K = ks*32 + (lane>=16?8:0) + (j&7) + (j>=8?16:0)]
//                                        [N = nt*16 + (lane&15)]
// Linear index = (((kl*NT + nt)*KS + ks)*32 + lane)*16 + j
// ---------------------------------------------------------------------------
__global__ void prep_w1(const float* __restrict__ W1, __bf16* __restrict__ dst) {
  int idx = blockIdx.x * 256 + threadIdx.x;
  if (idx >= (int)W1_HALVES) return;
  int j = idx & 15, lane = (idx >> 4) & 31;
  int t = idx >> 9;
  int ks = t % KS1; t /= KS1;
  int nt = t % NT1; t /= NT1;
  int kl = t;                               // k*NL + l, 0..7
  int kk = ks * 32 + ((lane >> 4) << 3) + (j & 7) + ((j >> 3) << 4);
  int n  = nt * 16 + (lane & 15);
  float v = (kk < L_N) ? W1[((size_t)kl * L_N + kk) * DFF_N + n] : 0.0f;
  dst[idx] = (__bf16)v;
}

__global__ void prep_w2(const float* __restrict__ W2, __bf16* __restrict__ dst) {
  int idx = blockIdx.x * 256 + threadIdx.x;
  if (idx >= (int)W2_HALVES) return;
  int j = idx & 15, lane = (idx >> 4) & 31;
  int t = idx >> 9;
  int ks = t % KS2; t /= KS2;
  int nt = t % NT2; t /= NT2;
  int kl = t;
  int kk = ks * 32 + ((lane >> 4) << 3) + (j & 7) + ((j >> 3) << 4);   // < 1024 always
  int n  = nt * 16 + (lane & 15);              // < 336 always
  dst[idx] = (__bf16)W2[((size_t)kl * DFF_N + kk) * L_N + n];
}

__global__ void prep_wh(const float* __restrict__ Wh, __bf16* __restrict__ dst) {
  int idx = blockIdx.x * 256 + threadIdx.x;
  if (idx >= (int)WH_HALVES) return;
  int j = idx & 15, lane = (idx >> 4) & 31;
  int t = idx >> 9;
  int ks = t % KS1; t /= KS1;
  int nt = t % NTH; t /= NTH;
  int k  = t;
  int kk = ks * 32 + ((lane >> 4) << 3) + (j & 7) + ((j >> 3) << 4);
  int n  = nt * 16 + (lane & 15);              // < 96 always
  float v = (kk < L_N) ? Wh[((size_t)k * L_N + kk) * O_N + n] : 0.0f;
  dst[idx] = (__bf16)v;
}

// ---------------------------------------------------------------------------
// Main fused kernel: one workgroup = one channel x ALL 32 batch rows (M=32),
// so every B fragment fetched from L2 feeds TWO WMMAs (two 16-row A tiles).
// 8 waves; WMMA N-tiles strided across waves (wave-uniform -> EXEC all-1s).
// LDS: z 32x356 f32 (45.5KB) + h 32x1032 bf16 (66KB) ~= 112KB (< 320KB/WGP).
// ---------------------------------------------------------------------------
__global__ __launch_bounds__(256) void tsmixer_main(
    const float* __restrict__ x, const float* __restrict__ rev_w,
    const float* __restrict__ rev_b, const float* __restrict__ b1,
    const float* __restrict__ b2, const float* __restrict__ bh,
    const int* __restrict__ assign, const __bf16* __restrict__ w1s,
    const __bf16* __restrict__ w2s, const __bf16* __restrict__ whs,
    float* __restrict__ out) {
  __shared__ float  zsm[32 * ZS];   // residual stream, f32
  __shared__ __bf16 hsm[32 * HS];   // hidden activation, bf16
  __shared__ float  s_mean[32], s_std[32];

  const int c    = blockIdx.x;       // channel
  const int tid  = threadIdx.x;
  const int lane = tid & 31;
  const int wv   = tid >> 5;
  const int k    = assign[c];
  const float rw = rev_w[c];
  const float rb = rev_b[c];

  // ---- RevIN stats + normalize (row = tid/8, 8 lanes per row) ----
  {
    const int r = tid >> 3, sub = tid & 7;
    const float* xr = x + ((size_t)r * C_N + c) * L_N;
    float vals[42];
    float s = 0.f, ss = 0.f;
#pragma unroll
    for (int t = 0; t < 42; ++t) {
      float v = xr[sub + 8 * t];
      vals[t] = v; s += v; ss += v * v;
    }
#pragma unroll
    for (int off = 4; off >= 1; off >>= 1) {     // reduce over 8-lane groups
      s  += __shfl_xor(s,  off, 32);
      ss += __shfl_xor(ss, off, 32);
    }
    float mean = s * (1.0f / L_N);
    float var  = ss * (1.0f / L_N) - mean * mean;
    float stdv = sqrtf(var + 1e-5f);
    float rstd = 1.0f / stdv;
    if (sub == 0) { s_mean[r] = mean; s_std[r] = stdv; }
#pragma unroll
    for (int t = 0; t < 42; ++t)
      zsm[r * ZS + sub + 8 * t] = (vals[t] - mean) * rstd * rw + rb;
    zsm[r * ZS + 336 + sub]     = 0.0f;          // zero-pad K to 352
    zsm[r * ZS + 336 + 8 + sub] = 0.0f;
  }
  __syncthreads();

  const int m   = lane & 15;                     // A-matrix row for this lane
  const int o1  = (lane >> 4) << 3;              // A K-offset: 0 or 8
  const int nc  = lane & 15;                     // D-tile column
  const int mrb = (lane >> 4) << 3;              // D-tile row base: 0 or 8

  const float* za0 = zsm + m * ZS + o1;          // A tile 0: rows 0-15
  const float* za1 = zsm + (m + 16) * ZS + o1;   // A tile 1: rows 16-31

  for (int layer = 0; layer < NL_N; ++layer) {
    const int kl = k * NL_N + layer;

    // ---- GEMM1: h = relu(z @ W1 + b1)  [32 x 1024] ----
    const __bf16* w1b = w1s + (size_t)kl * NT1 * (KS1 * 512) + lane * 16;
    for (int nt = wv; nt < NT1; nt += 8) {
      v8f acc0 = {0.f, 0.f, 0.f, 0.f, 0.f, 0.f, 0.f, 0.f};
      v8f acc1 = {0.f, 0.f, 0.f, 0.f, 0.f, 0.f, 0.f, 0.f};
      const __bf16* bp = w1b + (size_t)nt * (KS1 * 512);
#pragma unroll
      for (int ks = 0; ks < KS1; ++ks) {
        v16bf bf = *(const v16bf*)(bp + ks * 512);   // one B load ...
        v16bf a0 = loadA_f32(za0 + ks * 32);
        v16bf a1 = loadA_f32(za1 + ks * 32);
        acc0 = wmma_bf16(a0, bf, acc0);              // ... two WMMAs
        acc1 = wmma_bf16(a1, bf, acc1);
      }
      const int n = nt * 16 + nc;
      const float bias = b1[(size_t)kl * DFF_N + n];
#pragma unroll
      for (int i = 0; i < 8; ++i) {
        float v0 = acc0[i] + bias; v0 = v0 > 0.f ? v0 : 0.f;
        float v1 = acc1[i] + bias; v1 = v1 > 0.f ? v1 : 0.f;
        hsm[(mrb + i) * HS + n]      = (__bf16)v0;
        hsm[(mrb + i + 16) * HS + n] = (__bf16)v1;
      }
    }
    __syncthreads();

    // ---- GEMM2: z += h @ W2 + b2  [32 x 336] ----
    const __bf16* w2b = w2s + (size_t)kl * NT2 * (KS2 * 512) + lane * 16;
    const __bf16* ha0 = hsm + m * HS + o1;
    const __bf16* ha1 = hsm + (m + 16) * HS + o1;
    for (int nt = wv; nt < NT2; nt += 8) {
      v8f acc0 = {0.f, 0.f, 0.f, 0.f, 0.f, 0.f, 0.f, 0.f};
      v8f acc1 = {0.f, 0.f, 0.f, 0.f, 0.f, 0.f, 0.f, 0.f};
      const __bf16* bp = w2b + (size_t)nt * (KS2 * 512);
#pragma unroll
      for (int ks = 0; ks < KS2; ++ks) {
        v16bf bf = *(const v16bf*)(bp + ks * 512);
        v16bf a0 = loadA_bf16(ha0 + ks * 32);
        v16bf a1 = loadA_bf16(ha1 + ks * 32);
        acc0 = wmma_bf16(a0, bf, acc0);
        acc1 = wmma_bf16(a1, bf, acc1);
      }
      const int n = nt * 16 + nc;
      const float bias = b2[(size_t)kl * L_N + n];
#pragma unroll
      for (int i = 0; i < 8; ++i) {
        const int mi = mrb + i;
        zsm[mi * ZS + n]        = zsm[mi * ZS + n]        + acc0[i] + bias;
        zsm[(mi + 16) * ZS + n] = zsm[(mi + 16) * ZS + n] + acc1[i] + bias;
      }
    }
    __syncthreads();
  }

  // ---- Head: y = z @ Wh + bh, RevIN denorm, transpose to [B,O,C] ----
  const __bf16* whb = whs + (size_t)k * NTH * (KS1 * 512) + lane * 16;
  const float   rwi = 1.0f / rw;
  for (int nt = wv; nt < NTH; nt += 8) {
    v8f acc0 = {0.f, 0.f, 0.f, 0.f, 0.f, 0.f, 0.f, 0.f};
    v8f acc1 = {0.f, 0.f, 0.f, 0.f, 0.f, 0.f, 0.f, 0.f};
    const __bf16* bp = whb + (size_t)nt * (KS1 * 512);
#pragma unroll
    for (int ks = 0; ks < KS1; ++ks) {
      v16bf bf = *(const v16bf*)(bp + ks * 512);
      v16bf a0 = loadA_f32(za0 + ks * 32);
      v16bf a1 = loadA_f32(za1 + ks * 32);
      acc0 = wmma_bf16(a0, bf, acc0);
      acc1 = wmma_bf16(a1, bf, acc1);
    }
    const int n = nt * 16 + nc;
    const float bias = bh[(size_t)k * O_N + n];
#pragma unroll
    for (int i = 0; i < 8; ++i) {
      const int mi = mrb + i;
      float y0 = acc0[i] + bias;
      float y1 = acc1[i] + bias;
      y0 = (y0 - rb) * rwi * s_std[mi]      + s_mean[mi];
      y1 = (y1 - rb) * rwi * s_std[mi + 16] + s_mean[mi + 16];
      out[((size_t)mi        * O_N + n) * C_N + c] = y0;
      out[((size_t)(mi + 16) * O_N + n) * C_N + c] = y1;
    }
  }
}

// ---------------------------------------------------------------------------
extern "C" void kernel_launch(void* const* d_in, const int* in_sizes, int n_in,
                              void* d_out, int out_size, void* d_ws, size_t ws_size,
                              hipStream_t stream) {
  const float* x     = (const float*)d_in[0];
  const float* rev_w = (const float*)d_in[1];
  const float* rev_b = (const float*)d_in[2];
  const float* W1    = (const float*)d_in[3];
  const float* b1    = (const float*)d_in[4];
  const float* W2    = (const float*)d_in[5];
  const float* b2    = (const float*)d_in[6];
  const float* Wh    = (const float*)d_in[7];
  const float* bh    = (const float*)d_in[8];
  const int*   assign = (const int*)d_in[9];
  float* out = (float*)d_out;

  // Scratch layout: swizzled bf16 weights (needs ~11.6 MB of d_ws)
  __bf16* w1s = (__bf16*)d_ws;
  __bf16* w2s = w1s + W1_HALVES;
  __bf16* whs = w2s + W2_HALVES;

  prep_w1<<<(int)((W1_HALVES + 255) / 256), 256, 0, stream>>>(W1, w1s);
  prep_w2<<<(int)((W2_HALVES + 255) / 256), 256, 0, stream>>>(W2, w2s);
  prep_wh<<<(int)((WH_HALVES + 255) / 256), 256, 0, stream>>>(Wh, whs);

  tsmixer_main<<<C_N, 256, 0, stream>>>(x, rev_w, rev_b, b1, b2, bh, assign,
                                        w1s, w2s, whs, out);
}
